// WindowAttention_85375359910204
// MI455X (gfx1250) — matrix operand order repaired and checked by reading, not compiled, not run on hardware
//
#include <hip/hip_runtime.h>

#define BQ   128      // windows
#define NT   256      // tokens per window
#define NH   8        // heads
#define HDm  64       // head dim
#define DIMm 512
#define QKVm (3*DIMm)
#define LDSTR 264     // padded LDS row stride (floats)

typedef __bf16 bf16;
typedef __attribute__((ext_vector_type(16))) __bf16 v16bf;
typedef __attribute__((ext_vector_type(8)))  __bf16 v8bf;
typedef __attribute__((ext_vector_type(8)))  float  v8f;
typedef __attribute__((ext_vector_type(4)))  float  f32x4;
typedef __attribute__((ext_vector_type(4)))  unsigned u32x4;

static __device__ __forceinline__ unsigned short bfbits(bf16 h) {
  return __builtin_bit_cast(unsigned short, h);
}
static __device__ __forceinline__ bf16 f2bf(float f) {   // RNE
  union { float f; unsigned u; } a; a.f = f;
  unsigned r = a.u + 0x7FFFu + ((a.u >> 16) & 1u);
  return __builtin_bit_cast(bf16, (unsigned short)(r >> 16));
}
static __device__ __forceinline__ float bf2f(bf16 h) {
  union { unsigned u; float f; } a; a.u = ((unsigned)bfbits(h)) << 16;
  return a.f;
}
static __device__ __forceinline__ v16bf join8(v8bf a, v8bf b) {
  union { v16bf v; v8bf h[2]; } u; u.h[0] = a; u.h[1] = b; return u.v;
}

// A fragment (16x32): row-paired hi/lo planes, chunks at off1..+7 and off1+16..+23
static __device__ __forceinline__ void load_a_bf(const bf16* rp, int L, int off1,
                                                 v16bf& hi, v16bf& lo) {
  hi = join8(*(const v8bf*)(rp + off1),     *(const v8bf*)(rp + off1 + 16));
  lo = join8(*(const v8bf*)(rp + L + off1), *(const v8bf*)(rp + L + off1 + 16));
}
// B fragment (32x16): chunks at off..+7 and off+16..+23
static __device__ __forceinline__ void load_b_bf(const bf16* rp, int L, int off,
                                                 v16bf& hi, v16bf& lo) {
  hi = join8(*(const v8bf*)(rp + off),     *(const v8bf*)(rp + off + 16));
  lo = join8(*(const v8bf*)(rp + L + off), *(const v8bf*)(rp + L + off + 16));
}
// A fragment from LDS rows holding packed {lo[31:16], hi[15:0]} per element
static __device__ __forceinline__ void load_a_packed(const unsigned* rp, int off1,
                                                     v16bf& hi, v16bf& lo) {
  unsigned d[16];
  *(u32x4*)(d)      = *(const u32x4*)(rp + off1);
  *(u32x4*)(d + 4)  = *(const u32x4*)(rp + off1 + 4);
  *(u32x4*)(d + 8)  = *(const u32x4*)(rp + off1 + 16);
  *(u32x4*)(d + 12) = *(const u32x4*)(rp + off1 + 20);
  union { v16bf v; unsigned w[8]; } uh, ul;
#pragma unroll
  for (int t = 0; t < 8; ++t) {
    unsigned a = d[2 * t], b = d[2 * t + 1];
    uh.w[t] = (a & 0xFFFFu) | (b << 16);       // -> v_perm pattern
    ul.w[t] = (a >> 16) | (b & 0xFFFF0000u);
  }
  hi = uh.v; lo = ul.v;
}

static __device__ __forceinline__ v8f vzero() {
  v8f z;
#pragma unroll
  for (int i = 0; i < 8; ++i) z[i] = 0.f;
  return z;
}
// hi*hi + hi*lo + lo*hi : ~fp32-accurate product on the bf16 matrix engine
static __device__ __forceinline__ v8f wmma_split(v16bf ah, v16bf al, v16bf bhv,
                                                 v16bf blv, v8f c) {
  c = __builtin_amdgcn_wmma_f32_16x16x32_bf16(false, ah, false, bhv, (short)0, c, false, false);
  c = __builtin_amdgcn_wmma_f32_16x16x32_bf16(false, ah, false, blv, (short)0, c, false, false);
  c = __builtin_amdgcn_wmma_f32_16x16x32_bf16(false, al, false, bhv, (short)0, c, false, false);
  return c;
}

// ================= Kernel 0a: fp32 -> row-paired hi/lo bf16 planes (L=512) =================
__global__ __launch_bounds__(256) void cvt_kernel(const float* __restrict__ src,
                                                  bf16* __restrict__ dst, int total) {
  const int idx = (blockIdx.x * 256 + threadIdx.x) * 4;
  if (idx >= total) return;
  const f32x4 v = *(const f32x4*)(src + idx);
  const int row = idx >> 9, col = idx & 511;
  unsigned hw[2], lw[2];
#pragma unroll
  for (int t = 0; t < 2; ++t) {
    bf16 h0 = f2bf(v[2 * t]),               h1 = f2bf(v[2 * t + 1]);
    bf16 l0 = f2bf(v[2 * t] - bf2f(h0)),    l1 = f2bf(v[2 * t + 1] - bf2f(h1));
    hw[t] = (unsigned)bfbits(h0) | ((unsigned)bfbits(h1) << 16);
    lw[t] = (unsigned)bfbits(l0) | ((unsigned)bfbits(l1) << 16);
  }
  unsigned* dh = (unsigned*)(dst + (size_t)row * 1024 + col);
  dh[0] = hw[0]; dh[1] = hw[1];
  unsigned* dl = (unsigned*)(dst + (size_t)row * 1024 + 512 + col);
  dl[0] = lw[0]; dl[1] = lw[1];
}

// ====== Kernel 0b: bmap[h][i][j] = rpb_table[relidx(i,j)][h] + pe_b[h]  (2 MB, L2-resident) ======
__global__ __launch_bounds__(256) void bias_kernel(const float* __restrict__ rpb,
                                                   const float* __restrict__ pe_b,
                                                   float* __restrict__ bmap) {
  const int idx = blockIdx.x * 256 + threadIdx.x;  // h*65536 + i*256 + j
  const int h = idx >> 16;
  const int i = (idx >> 8) & 255;
  const int j = idx & 255;
  const int rel = ((i >> 4) - (j >> 4) + 15) * 31 + ((i & 15) - (j & 15) + 15);
  bmap[idx] = rpb[rel * NH + h] + pe_b[h];
}

// ============================ Kernel 1: fused QKV GEMM ============================
__global__ __launch_bounds__(256) void qkv_kernel(
    const bf16* __restrict__ x2, const bf16* __restrict__ w2,
    const float* __restrict__ bias, bf16* __restrict__ qs2,
    bf16* __restrict__ kb2, bf16* __restrict__ vt2)
{
  const int lane = threadIdx.x & 31;
  const int wave = threadIdx.x >> 5;
  const int wm = wave >> 1, wn = wave & 1;
  const int m0 = blockIdx.y * 128 + wm * 32;
  const int n0 = blockIdx.x * 128 + wn * 64;
  const int r = lane & 15, g = lane >> 4;

  v8f c[2][4];
#pragma unroll
  for (int i = 0; i < 2; ++i)
#pragma unroll
    for (int j = 0; j < 4; ++j) c[i][j] = vzero();

  for (int kc = 0; kc < DIMm; kc += 32) {
    v16bf ah[2], al[2], bh[4], bl[4];
#pragma unroll
    for (int mt = 0; mt < 2; ++mt)
      load_a_bf(x2 + (size_t)(m0 + 16 * mt + r) * 1024, 512, kc + 8 * g, ah[mt], al[mt]);
#pragma unroll
    for (int nt = 0; nt < 4; ++nt)
      load_b_bf(w2 + (size_t)(n0 + 16 * nt + r) * 1024, 512, kc + 8 * g, bh[nt], bl[nt]);
#pragma unroll
    for (int mt = 0; mt < 2; ++mt)
#pragma unroll
      for (int nt = 0; nt < 4; ++nt)
        c[mt][nt] = wmma_split(ah[mt], al[mt], bh[nt], bl[nt], c[mt][nt]);
  }

  const float scale = 0.125f;  // HD^-0.5
#pragma unroll
  for (int mt = 0; mt < 2; ++mt) {
#pragma unroll
    for (int nt = 0; nt < 4; ++nt) {
      const int col = n0 + 16 * nt + r;
      const float bval = bias[col];
      const int which = col >> 9;           // 0=q 1=k 2=v (uniform per tile)
      const int rem = col & 511;
      const int hh = rem >> 6, hd = rem & 63;
#pragma unroll
      for (int j = 0; j < 8; ++j) {
        const int m = m0 + 16 * mt + 8 * g + j;
        const int b = m >> 8, tok = m & 255;
        float v = c[mt][nt][j] + bval;
        if (which == 0) v *= scale;
        const bf16 hb = f2bf(v);
        const bf16 lb = f2bf(v - bf2f(hb));
        if (which == 0) {
          const size_t base = (((size_t)b * NH + hh) * NT + tok) * 128 + hd;
          qs2[base] = hb; qs2[base + 64] = lb;
        } else if (which == 1) {
          const size_t base = (((size_t)b * NH + hh) * NT + tok) * 128 + hd;
          kb2[base] = hb; kb2[base + 64] = lb;
        } else {
          const size_t base = (((size_t)b * NH + hh) * HDm + hd) * 512 + tok;
          vt2[base] = hb; vt2[base + 256] = lb;   // v transposed
        }
      }
    }
  }
}

// ================= Kernel 2: fused attention, one WG per (window, head) =================
__global__ __launch_bounds__(256) void attn_kernel(
    const bf16* __restrict__ qs2, const bf16* __restrict__ kb2,
    const bf16* __restrict__ vt2, const float* __restrict__ pe_w,
    const float* __restrict__ bmap, bf16* __restrict__ ao2)
{
  extern __shared__ float Sm[];            // [NT][LDSTR] score map: 264 KB
  const int bhid = blockIdx.x;
  const int b = bhid >> 3, h = bhid & 7;
  const int tid = threadIdx.x;
  const int lane = tid & 31, wave = tid >> 5;
  const int r = lane & 15, g = lane >> 4;
  const bf16* qbase = qs2 + (size_t)bhid * NT * 128;
  const bf16* kbase = kb2 + (size_t)bhid * NT * 128;
  const bf16* vbase = vt2 + (size_t)bhid * HDm * 512;
  const float* brow = bmap + (size_t)h * NT * NT;
  const int m0 = wave * 32;                // this wave owns query rows [m0, m0+32)

  // ---- Phase 1: S = (q*scale) @ k^T  (M=256, N=256, K=64) ----
  {
    v16bf qh[2][2], ql[2][2];
#pragma unroll
    for (int mt = 0; mt < 2; ++mt)
#pragma unroll
      for (int kk = 0; kk < 2; ++kk)
        load_a_bf(qbase + (size_t)(m0 + 16 * mt + r) * 128, 64, 32 * kk + 8 * g,
                  qh[mt][kk], ql[mt][kk]);

    for (int nb = 0; nb < NT; nb += 64) {
      v8f c[2][4];
#pragma unroll
      for (int i = 0; i < 2; ++i)
#pragma unroll
        for (int j = 0; j < 4; ++j) c[i][j] = vzero();
#pragma unroll
      for (int kk = 0; kk < 2; ++kk) {
        v16bf bhf[4], blf[4];
#pragma unroll
        for (int nt = 0; nt < 4; ++nt)
          load_b_bf(kbase + (size_t)(nb + 16 * nt + r) * 128, 64, 32 * kk + 8 * g,
                    bhf[nt], blf[nt]);
#pragma unroll
        for (int mt = 0; mt < 2; ++mt)
#pragma unroll
          for (int nt = 0; nt < 4; ++nt)
            c[mt][nt] = wmma_split(qh[mt][kk], ql[mt][kk], bhf[nt], blf[nt], c[mt][nt]);
      }
#pragma unroll
      for (int mt = 0; mt < 2; ++mt)
#pragma unroll
        for (int nt = 0; nt < 4; ++nt)
#pragma unroll
          for (int j = 0; j < 8; ++j)
            Sm[(m0 + 16 * mt + 8 * g + j) * LDSTR + nb + 16 * nt + r] = c[mt][nt][j];
    }
  }

  // Warm caches with this WG's v^T slice (64 KB): the serial conv/softmax phases
  // hide the HBM latency; global_prefetch does not touch LOADcnt.
  {
    const char* pv = (const char*)vbase;
    for (int off = tid * 128; off < HDm * 512 * 2; off += 256 * 128)
      __builtin_prefetch(pv + off, 0, 3);
  }
  __syncthreads();

  // ---- Phase 2: EPEG conv (15,1) along rows + precomputed bias map; column-per-thread,
  // 15-tap register ring -> in-place update with no cross-thread hazard.
  {
    const int j = tid;
    float w15[15];
#pragma unroll
    for (int t = 0; t < 15; ++t) w15[t] = pe_w[h * 15 + t];
    float win[15];
#pragma unroll
    for (int t = 0; t < 15; ++t) {
      const int i = t - 7;
      win[t] = (i >= 0) ? Sm[i * LDSTR + j] : 0.f;
    }
    for (int i = 0; i < NT; ++i) {
      float conv = 0.f;
#pragma unroll
      for (int t = 0; t < 15; ++t) conv += w15[t] * win[t];
      Sm[i * LDSTR + j] = win[7] + conv + brow[i * NT + j];   // coalesced bias load
#pragma unroll
      for (int t = 0; t < 14; ++t) win[t] = win[t + 1];
      const int nx = i + 8;
      win[14] = (nx < NT) ? Sm[nx * LDSTR + j] : 0.f;
    }
  }
  __syncthreads();

  // ---- Phase 3: softmax (row-per-thread); final pass writes packed {lo,hi} bf16
  // into the SAME dword slot just read -> in-place, hazard-free.
  {
    float* row = Sm + (size_t)tid * LDSTR;
    unsigned* urow = (unsigned*)row;
    float mx = -3.4e38f;
    for (int jj = 0; jj < NT; ++jj) mx = fmaxf(mx, row[jj]);
    float s = 0.f;
    for (int jj = 0; jj < NT; ++jj) s += __expf(row[jj] - mx);
    const float inv = 1.f / s;
    for (int jj = 0; jj < NT; ++jj) {
      const float e = __expf(row[jj] - mx) * inv;
      const bf16 hb = f2bf(e);
      const bf16 lb = f2bf(e - bf2f(hb));
      urow[jj] = (unsigned)bfbits(hb) | ((unsigned)bfbits(lb) << 16);
    }
  }
  __syncthreads();

  // ---- Phase 4: O = P @ v  (M=256, N=64, K=256); A unpacked from LDS, B = v^T planes ----
  {
    v8f c[2][4];
#pragma unroll
    for (int i = 0; i < 2; ++i)
#pragma unroll
      for (int j = 0; j < 4; ++j) c[i][j] = vzero();
    for (int kc = 0; kc < NT; kc += 32) {
      v16bf ah2[2], al2[2], bh2[4], bl2[4];
#pragma unroll
      for (int mt = 0; mt < 2; ++mt)
        load_a_packed((const unsigned*)(Sm + (size_t)(m0 + 16 * mt + r) * LDSTR),
                      kc + 8 * g, ah2[mt], al2[mt]);
#pragma unroll
      for (int nt = 0; nt < 4; ++nt)
        load_b_bf(vbase + (size_t)(16 * nt + r) * 512, 256, kc + 8 * g, bh2[nt], bl2[nt]);
#pragma unroll
      for (int mt = 0; mt < 2; ++mt)
#pragma unroll
        for (int nt = 0; nt < 4; ++nt)
          c[mt][nt] = wmma_split(ah2[mt], al2[mt], bh2[nt], bl2[nt], c[mt][nt]);
    }
#pragma unroll
    for (int mt = 0; mt < 2; ++mt)
#pragma unroll
      for (int nt = 0; nt < 4; ++nt)
#pragma unroll
        for (int j = 0; j < 8; ++j) {
          const int tok = m0 + 16 * mt + 8 * g + j;
          const float o = c[mt][nt][j];
          const bf16 hb = f2bf(o);
          const bf16 lb = f2bf(o - bf2f(hb));
          const size_t base = ((size_t)b * NT + tok) * 1024 + h * HDm + 16 * nt + r;
          ao2[base] = hb; ao2[base + 512] = lb;
        }
  }
}

// ============================ Kernel 3: output projection ============================
__global__ __launch_bounds__(256) void proj_kernel(
    const bf16* __restrict__ a2, const bf16* __restrict__ w2,
    const float* __restrict__ bias, float* __restrict__ out)
{
  const int lane = threadIdx.x & 31;
  const int wave = threadIdx.x >> 5;
  const int wm = wave >> 1, wn = wave & 1;
  const int m0 = blockIdx.y * 128 + wm * 32;
  const int n0 = blockIdx.x * 128 + wn * 64;
  const int r = lane & 15, g = lane >> 4;

  v8f c[2][4];
#pragma unroll
  for (int i = 0; i < 2; ++i)
#pragma unroll
    for (int j = 0; j < 4; ++j) c[i][j] = vzero();

  for (int kc = 0; kc < DIMm; kc += 32) {
    v16bf ah[2], al[2], bh[4], bl[4];
#pragma unroll
    for (int mt = 0; mt < 2; ++mt)
      load_a_bf(a2 + (size_t)(m0 + 16 * mt + r) * 1024, 512, kc + 8 * g, ah[mt], al[mt]);
#pragma unroll
    for (int nt = 0; nt < 4; ++nt)
      load_b_bf(w2 + (size_t)(n0 + 16 * nt + r) * 1024, 512, kc + 8 * g, bh[nt], bl[nt]);
#pragma unroll
    for (int mt = 0; mt < 2; ++mt)
#pragma unroll
      for (int nt = 0; nt < 4; ++nt)
        c[mt][nt] = wmma_split(ah[mt], al[mt], bh[nt], bl[nt], c[mt][nt]);
  }
#pragma unroll
  for (int mt = 0; mt < 2; ++mt)
#pragma unroll
    for (int nt = 0; nt < 4; ++nt) {
      const int col = n0 + 16 * nt + r;
      const float bval = bias[col];
#pragma unroll
      for (int j = 0; j < 8; ++j) {
        const int m = m0 + 16 * mt + 8 * g + j;
        out[(size_t)m * DIMm + col] = c[mt][nt][j] + bval;
      }
    }
}

extern "C" void kernel_launch(void* const* d_in, const int* in_sizes, int n_in,
                              void* d_out, int out_size, void* d_ws, size_t ws_size,
                              hipStream_t stream) {
  (void)in_sizes; (void)n_in; (void)out_size; (void)ws_size;
  const float* x      = (const float*)d_in[0];
  const float* qkv_w  = (const float*)d_in[1];
  const float* qkv_b  = (const float*)d_in[2];
  const float* pe_w   = (const float*)d_in[3];
  const float* pe_b   = (const float*)d_in[4];
  const float* rpb    = (const float*)d_in[5];
  const float* proj_w = (const float*)d_in[6];
  const float* proj_b = (const float*)d_in[7];
  float* out = (float*)d_out;

  // workspace layout (hi/lo bf16 planes == same bytes as fp32)
  char* p = (char*)d_ws;
  bf16* x2  = (bf16*)p;  p += (size_t)32768 * 1024 * 2;        // 64 MB
  bf16* qw2 = (bf16*)p;  p += (size_t)QKVm * 1024 * 2;         //  3 MB
  bf16* pw2 = (bf16*)p;  p += (size_t)DIMm * 1024 * 2;         //  1 MB
  bf16* qs2 = (bf16*)p;  p += (size_t)BQ * NH * NT * 128 * 2;  // 64 MB
  bf16* kb2 = (bf16*)p;  p += (size_t)BQ * NH * NT * 128 * 2;  // 64 MB
  bf16* vt2 = (bf16*)p;  p += (size_t)BQ * NH * HDm * 512 * 2; // 64 MB
  bf16* ao2 = (bf16*)p;  p += (size_t)32768 * 1024 * 2;        // 64 MB
  float* bmap = (float*)p;                                     //  2 MB

  dim3 blk(256);
  cvt_kernel<<<dim3((32768 * 512) / 1024), blk, 0, stream>>>(x, x2, 32768 * 512);
  cvt_kernel<<<dim3((QKVm * 512) / 1024), blk, 0, stream>>>(qkv_w, qw2, QKVm * 512);
  cvt_kernel<<<dim3((DIMm * 512) / 1024), blk, 0, stream>>>(proj_w, pw2, DIMm * 512);
  bias_kernel<<<dim3((NH * NT * NT) / 256), blk, 0, stream>>>(rpb, pe_b, bmap);

  qkv_kernel<<<dim3(QKVm / 128, (BQ * NT) / 128), blk, 0, stream>>>(
      x2, qw2, qkv_b, qs2, kb2, vt2);
  attn_kernel<<<dim3(BQ * NH), blk, NT * LDSTR * sizeof(float), stream>>>(
      qs2, kb2, vt2, pe_w, bmap, ao2);
  proj_kernel<<<dim3(DIMm / 128, (BQ * NT) / 128), blk, 0, stream>>>(
      ao2, pw2, proj_b, out);
}
